// Attention_86242943303884
// MI455X (gfx1250) — matrix operand order repaired and checked by reading, not compiled, not run on hardware
//
#include <hip/hip_runtime.h>
#include <hip/hip_bf16.h>

#define BATCH   2
#define SEQ     2048
#define DMODEL  2048
#define NHEADS  16
#define KVHEADS 4
#define HDIM    128
#define NREP    (NHEADS / KVHEADS)

typedef __attribute__((ext_vector_type(16))) _Float16 v16h;
typedef __attribute__((ext_vector_type(8)))  _Float16 v8h;
typedef __attribute__((ext_vector_type(8)))  float    v8f;

static __device__ __forceinline__ v8f wmma_f16(v16h a, v16h b, v8f c) {
  // D = A(16x32 f16) x B(32x16 f16) + C(16x16 f32)
  return __builtin_amdgcn_wmma_f32_16x16x32_f16(false, a, false, b, (short)0, c,
                                                false, false);
}

// A-fragment (16x32 f16): M = lane&15; K split {0..15} / {16..31}
// across lane halves. Caller pre-offsets p by (lane>>4)*16 halves.
static __device__ __forceinline__ v16h load_a_frag(const _Float16* p) {
  v8h lo = *(const v8h*)(p);
  v8h hi = *(const v8h*)(p + 8);
  v16h a;
#pragma unroll
  for (int i = 0; i < 8; ++i) { a[i] = lo[i]; a[i + 8] = hi[i]; }
  return a;
}

// ---------------- gfx1250 async global->LDS copy (ASYNCcnt-tracked) ------------
// Each call: every lane copies 16 bytes global -> LDS. Emitted via inline asm:
// the clang builtin's params are in the (unspellable) __device__/__shared__
// language address spaces, so we go straight to the ISA. VDST = LDS byte
// address (low 32 bits of the flat shared pointer), VADDR = 64-bit global addr.
static __device__ __forceinline__ void async_cp16(const _Float16* g, _Float16* l) {
  unsigned lofs = (unsigned)(unsigned long long)l;  // flat LDS addr[31:0] = LDS offset
  asm volatile("global_load_async_to_lds_b128 %0, %1, off" ::"v"(lofs),
               "v"((unsigned long long)g)
               : "memory");
}

static __device__ __forceinline__ void wait_async0() {
#if __has_builtin(__builtin_amdgcn_s_wait_asynccnt)
  __builtin_amdgcn_s_wait_asynccnt(0);
#else
  asm volatile("s_wait_asynccnt 0x0" ::: "memory");
#endif
  asm volatile("" ::: "memory");
}

static __device__ __forceinline__ void wait_async32() {
#if __has_builtin(__builtin_amdgcn_s_wait_asynccnt)
  __builtin_amdgcn_s_wait_asynccnt(32);
#else
  asm volatile("s_wait_asynccnt 0x20" ::: "memory");
#endif
  asm volatile("" ::: "memory");
}

#define SK_PAD 16  // K tile row stride 144 halves = 288B (32B aligned, bank-spread)
#define SV_PAD 16  // V tile row stride 48 halves = 96B (32B aligned)

// Stage one 32-key chunk: K (32 rows x 256B) + V (128 d-rows x 64B) = 32 async ops.
static __device__ __forceinline__ void stage_kv(
    const _Float16* kbase, const _Float16* vbase, int c0, int lane,
    _Float16 (*sk)[HDIM + SK_PAD], _Float16 (*sv)[32 + SV_PAD]) {
  {
    const _Float16* g = kbase + (size_t)(c0 + lane) * HDIM;
    _Float16* l = sk[lane];
#pragma unroll
    for (int seg = 0; seg < 16; ++seg) {  // 16 x (32 lanes x 16B) = 8KB
      async_cp16(g, l);
      g += 8;
      l += 8;
    }
  }
#pragma unroll
  for (int rb = 0; rb < 4; ++rb) {
    const _Float16* g = vbase + (size_t)(rb * 32 + lane) * SEQ + c0;
    _Float16* l = sv[rb * 32 + lane];
#pragma unroll
    for (int seg = 0; seg < 4; ++seg) {  // 4 rb x 4 seg x 512B = 8KB
      async_cp16(g, l);
      g += 8;
      l += 8;
    }
  }
}

// ---------------- elementwise conversion kernels ----------------

__global__ void cvt_f32_f16(const float* __restrict__ src,
                            _Float16* __restrict__ dst, int n) {
  int i = blockIdx.x * blockDim.x + threadIdx.x;
  if (i < n) dst[i] = (_Float16)src[i];
}

// src [K,N] f32 -> dst [N,K] f16  (weights transposed so B-fragments are contiguous)
__global__ void cvt_transpose(const float* __restrict__ src,
                              _Float16* __restrict__ dst, int K, int N) {
  int i = blockIdx.x * blockDim.x + threadIdx.x;
  if (i >= K * N) return;
  int k = i % K;
  int n = i / K;
  dst[(size_t)n * K + k] = (_Float16)src[(size_t)k * N + n];
}

// RoPE + layout change: src f32 [B,S,H,HDIM] -> dst f16 [B,H,S,HDIM]
__global__ void rope_cvt(const float* __restrict__ src,
                         _Float16* __restrict__ dst, int H) {
  int i = blockIdx.x * blockDim.x + threadIdx.x;  // pair index
  int total = BATCH * SEQ * H * (HDIM / 2);
  if (i >= total) return;
  int d2 = i & 63;  // HDIM/2 = 64
  int t  = i >> 6;  // (b*SEQ+s)*H + h
  int h  = t % H;
  int t2 = t / H;
  int s  = t2 % SEQ;
  int b  = t2 / SEQ;
  float inv = __powf(10000.0f, -(float)d2 / 64.0f);
  float ang = (float)s * inv;
  float sn, cs;
  __sincosf(ang, &sn, &cs);
  size_t si = (((size_t)b * SEQ + s) * H + h) * HDIM + 2 * d2;
  float x0 = src[si], x1 = src[si + 1];
  size_t di = (((size_t)b * H + h) * SEQ + s) * HDIM + 2 * d2;
  dst[di]     = (_Float16)(x0 * cs - x1 * sn);
  dst[di + 1] = (_Float16)(x0 * sn + x1 * cs);
}

// V: src f32 [B,S,KVH,HDIM] -> dst f16 [B,KVH,HDIM,S] (d-major so PV B-frags are contiguous)
__global__ void v_transpose_cvt(const float* __restrict__ src,
                                _Float16* __restrict__ dst) {
  int i = blockIdx.x * blockDim.x + threadIdx.x;
  int total = BATCH * SEQ * KVHEADS * HDIM;
  if (i >= total) return;
  int d  = i % HDIM;
  int t  = i / HDIM;
  int h  = t % KVHEADS;
  int t2 = t / KVHEADS;
  int s  = t2 % SEQ;
  int b  = t2 / SEQ;
  dst[(((size_t)b * KVHEADS + h) * HDIM + d) * SEQ + s] = (_Float16)src[i];
}

// ---------------- WMMA GEMM: C[M,N] f32 = A[M,K] f16 x BT[N,K] f16 ----------------
// One wave computes a 16x64 tile (A-fragment reused across 4 N-tiles).
__global__ __launch_bounds__(256) void gemm_f16_wmma(
    const _Float16* __restrict__ A, const _Float16* __restrict__ BT,
    float* __restrict__ C, int M, int N, int K) {
  const int lane   = threadIdx.x & 31;
  const int wave   = blockIdx.x * (blockDim.x >> 5) + (threadIdx.x >> 5);
  const int tilesN = N >> 6;
  const int tm = (wave / tilesN) << 4;
  const int tn = (wave % tilesN) << 6;
  if (tm >= M) return;

  const int lid  = lane & 15;
  const int half = lane >> 4;

  const _Float16* arow = A + (size_t)(tm + lid) * K + (half << 4);
  const _Float16* brow = BT + (size_t)(tn + lid) * K + (half << 4);
  const size_t bstride = (size_t)16 * K;  // next 16-wide N tile

  v8f acc[4] = {v8f{}, v8f{}, v8f{}, v8f{}};

  for (int kk = 0; kk < K; kk += 32) {
    __builtin_prefetch(arow + kk + 256, 0, 0);  // global_prefetch_b8
    __builtin_prefetch(brow + kk + 256, 0, 0);
    v16h a = load_a_frag(arow + kk);
#pragma unroll
    for (int j = 0; j < 4; ++j) {
      v16h b = *(const v16h*)(brow + (size_t)j * bstride + kk);
      acc[j] = wmma_f16(a, b, acc[j]);
    }
  }

  const int row0 = tm + (half << 3);
#pragma unroll
  for (int j = 0; j < 4; ++j)
#pragma unroll
    for (int r = 0; r < 8; ++r)
      C[(size_t)(row0 + r) * N + tn + j * 16 + lid] = acc[j][r];
}

// ------- flash attention: one wave = one (b, h, 16-query block), -------
// ------- async double-buffered K/V tiles in LDS (ASYNCcnt pipeline) -------
__global__ __launch_bounds__(32) void flash_attn_wmma(
    const _Float16* __restrict__ q16,   // [B,NH,S,HD]
    const _Float16* __restrict__ k16,   // [B,KVH,S,HD]
    const _Float16* __restrict__ vT16,  // [B,KVH,HD,S]
    _Float16* __restrict__ attn) {      // [B*S, NH*HD]
  __shared__ _Float16 sK[2][32][HDIM + SK_PAD];   // 18 KB
  __shared__ _Float16 sV[2][HDIM][32 + SV_PAD];   // 24 KB
  __shared__ float sP[16][33];                    // 2.1 KB

  const int lane = threadIdx.x & 31;
  const int lid  = lane & 15;
  const int half = lane >> 4;

  const int job = blockIdx.x;  // b*NH*(S/16)
  const int qb  = job % (SEQ / 16);
  const int h   = (job / (SEQ / 16)) % NHEADS;
  const int b   = job / ((SEQ / 16) * NHEADS);
  const int hk  = h / NREP;
  const int qrow0 = qb * 16;

  // Q A-fragments for the whole 16x128 block (4 K-steps of 32), register resident
  const _Float16* qrow =
      q16 + (((size_t)b * NHEADS + h) * SEQ + qrow0 + lid) * HDIM + half * 16;
  v16h aq[4];
#pragma unroll
  for (int kk = 0; kk < 4; ++kk) aq[kk] = load_a_frag(qrow + kk * 32);

  const _Float16* kbase = k16 + ((size_t)b * KVHEADS + hk) * SEQ * HDIM;
  const _Float16* vbase = vT16 + ((size_t)b * KVHEADS + hk) * (size_t)HDIM * SEQ;

  v8f acc[8];
#pragma unroll
  for (int j = 0; j < 8; ++j) acc[j] = v8f{};
  float mrow[8], lrow[8];
#pragma unroll
  for (int r = 0; r < 8; ++r) { mrow[r] = -1.0e30f; lrow[r] = 0.0f; }

  const float SC = 0.08838834764831845f;  // 1/sqrt(128)
  const int lastrow = qrow0 + 15;
  const int nch = (lastrow >> 5) + 1;

  stage_kv(kbase, vbase, 0, lane, sK[0], sV[0]);

  for (int ci = 0; ci < nch; ++ci) {
    const int c0 = ci << 5;
    _Float16 (*sk)[HDIM + SK_PAD] = sK[ci & 1];
    _Float16 (*sv)[32 + SV_PAD]   = sV[ci & 1];

    // prefetch next chunk into the other buffer, then wait for this chunk
    if (ci + 1 < nch) {
      stage_kv(kbase, vbase, c0 + 32, lane, sK[(ci + 1) & 1], sV[(ci + 1) & 1]);
      wait_async32();  // async loads retire in order: <=32 outstanding => chunk ci resident
    } else {
      wait_async0();
    }

    // ---- scores: two 16x16 tiles from LDS, reduction over HD=128 ----
    v8f s0 = v8f{}, s1 = v8f{};
#pragma unroll
    for (int kk = 0; kk < 4; ++kk) {
      v16h b0 = *(const v16h*)&sk[lid][half * 16 + kk * 32];
      s0 = wmma_f16(aq[kk], b0, s0);
      v16h b1 = *(const v16h*)&sk[16 + lid][half * 16 + kk * 32];
      s1 = wmma_f16(aq[kk], b1, s1);
    }

    // ---- causal mask + scale + online softmax (rows live in lane halves) ----
    float mnew[8];
#pragma unroll
    for (int r = 0; r < 8; ++r) {
      int row = qrow0 + r + half * 8;
      float v0 = s0[r] * SC;
      if (c0 + lid > row) v0 = -1.0e30f;
      float v1 = s1[r] * SC;
      if (c0 + 16 + lid > row) v1 = -1.0e30f;
      s0[r] = v0;
      s1[r] = v1;
      float m = fmaxf(v0, v1);
      m = fmaxf(m, __shfl_xor(m, 1));
      m = fmaxf(m, __shfl_xor(m, 2));
      m = fmaxf(m, __shfl_xor(m, 4));
      m = fmaxf(m, __shfl_xor(m, 8));
      mnew[r] = fmaxf(mrow[r], m);
    }
#pragma unroll
    for (int r = 0; r < 8; ++r) {
      float alpha = __expf(mrow[r] - mnew[r]);
      float e0 = __expf(s0[r] - mnew[r]);
      float e1 = __expf(s1[r] - mnew[r]);
      float ps = e0 + e1;
      ps += __shfl_xor(ps, 1);
      ps += __shfl_xor(ps, 2);
      ps += __shfl_xor(ps, 4);
      ps += __shfl_xor(ps, 8);
      lrow[r] = lrow[r] * alpha + ps;
      mrow[r] = mnew[r];
      sP[r + half * 8][lid]      = e0;  // C-layout -> LDS
      sP[r + half * 8][16 + lid] = e1;
#pragma unroll
      for (int j = 0; j < 8; ++j) acc[j][r] *= alpha;
    }
    __syncthreads();

    // ---- re-pack P from LDS into A-fragment layout (16x32) ----
    v16h ap;
#pragma unroll
    for (int e = 0; e < 8; ++e) {
      ap[e]     = (_Float16)sP[lid][half * 16 + e];
      ap[e + 8] = (_Float16)sP[lid][half * 16 + 8 + e];
    }
    __syncthreads();

    // ---- P(16x32) x V(32x16) from LDS for each of 8 head-dim tiles ----
#pragma unroll
    for (int j = 0; j < 8; ++j) {
      v16h bv = *(const v16h*)&sv[j * 16 + lid][half * 16];
      acc[j] = wmma_f16(ap, bv, acc[j]);
    }
  }

  // ---- epilogue: divide by row sums, write [B*S, NH*HD] f16 ----
#pragma unroll
  for (int r = 0; r < 8; ++r) {
    int row = qrow0 + r + half * 8;
    float inv = 1.0f / lrow[r];
    size_t base = (size_t)(b * SEQ + row) * (NHEADS * HDIM) + h * HDIM;
#pragma unroll
    for (int j = 0; j < 8; ++j)
      attn[base + j * 16 + lid] = (_Float16)(acc[j][r] * inv);
  }
}

// ---------------- host-side launch ----------------

extern "C" void kernel_launch(void* const* d_in, const int* in_sizes, int n_in,
                              void* d_out, int out_size, void* d_ws, size_t ws_size,
                              hipStream_t stream) {
  (void)in_sizes; (void)n_in; (void)out_size; (void)ws_size;

  const float* x  = (const float*)d_in[0];
  // d_in[1] = causal mask, applied analytically in the flash kernel
  const float* Wq = (const float*)d_in[2];
  const float* Wk = (const float*)d_in[3];
  const float* Wv = (const float*)d_in[4];
  const float* Wo = (const float*)d_in[5];
  float* out = (float*)d_out;

  char* w = (char*)d_ws;
  auto alloc = [&](size_t bytes) -> void* {
    void* p = (void*)w;
    w += (bytes + 255) & ~(size_t)255;
    return p;
  };

  const int MROWS = BATCH * SEQ;  // 4096
  _Float16* x16    = (_Float16*)alloc((size_t)MROWS * DMODEL * 2);
  _Float16* WqT    = (_Float16*)alloc((size_t)DMODEL * DMODEL * 2);
  _Float16* WkT    = (_Float16*)alloc((size_t)DMODEL * (KVHEADS * HDIM) * 2);
  _Float16* WvT    = (_Float16*)alloc((size_t)DMODEL * (KVHEADS * HDIM) * 2);
  _Float16* WoT    = (_Float16*)alloc((size_t)DMODEL * DMODEL * 2);
  float*    Qf     = (float*)alloc((size_t)MROWS * DMODEL * 4);
  float*    Kf     = (float*)alloc((size_t)MROWS * (KVHEADS * HDIM) * 4);
  float*    Vf     = (float*)alloc((size_t)MROWS * (KVHEADS * HDIM) * 4);
  _Float16* q16    = (_Float16*)alloc((size_t)BATCH * NHEADS * SEQ * HDIM * 2);
  _Float16* k16    = (_Float16*)alloc((size_t)BATCH * KVHEADS * SEQ * HDIM * 2);
  _Float16* vT16   = (_Float16*)alloc((size_t)BATCH * KVHEADS * HDIM * SEQ * 2);
  _Float16* attn16 = (_Float16*)alloc((size_t)MROWS * DMODEL * 2);

  auto grid1d = [](int n) { return (n + 255) / 256; };

  // f32 -> f16 activations, f32 -> f16 transposed weights
  {
    int n = MROWS * DMODEL;
    cvt_f32_f16<<<grid1d(n), 256, 0, stream>>>(x, x16, n);
  }
  cvt_transpose<<<grid1d(DMODEL * DMODEL), 256, 0, stream>>>(Wq, WqT, DMODEL, DMODEL);
  cvt_transpose<<<grid1d(DMODEL * KVHEADS * HDIM), 256, 0, stream>>>(Wk, WkT, DMODEL, KVHEADS * HDIM);
  cvt_transpose<<<grid1d(DMODEL * KVHEADS * HDIM), 256, 0, stream>>>(Wv, WvT, DMODEL, KVHEADS * HDIM);
  cvt_transpose<<<grid1d(DMODEL * DMODEL), 256, 0, stream>>>(Wo, WoT, DMODEL, DMODEL);

  auto gemm = [&](const _Float16* A, const _Float16* BT, float* C, int M, int N,
                  int K) {
    int waves = (M / 16) * (N / 64);
    gemm_f16_wmma<<<waves / 8, 256, 0, stream>>>(A, BT, C, M, N, K);
  };

  // projections
  gemm(x16, WqT, Qf, MROWS, DMODEL, DMODEL);
  gemm(x16, WkT, Kf, MROWS, KVHEADS * HDIM, DMODEL);
  gemm(x16, WvT, Vf, MROWS, KVHEADS * HDIM, DMODEL);

  // RoPE + layout changes
  rope_cvt<<<grid1d(BATCH * SEQ * NHEADS * (HDIM / 2)), 256, 0, stream>>>(Qf, q16, NHEADS);
  rope_cvt<<<grid1d(BATCH * SEQ * KVHEADS * (HDIM / 2)), 256, 0, stream>>>(Kf, k16, KVHEADS);
  v_transpose_cvt<<<grid1d(BATCH * SEQ * KVHEADS * HDIM), 256, 0, stream>>>(Vf, vT16);

  // flash attention: one wave per (b, head, 16-query block)
  {
    int jobs = BATCH * NHEADS * (SEQ / 16);  // 4096
    flash_attn_wmma<<<jobs, 32, 0, stream>>>(q16, k16, vT16, attn16);
  }

  // output projection
  gemm(attn16, WoT, out, MROWS, DMODEL, DMODEL);
}
